// TaskConcept_14723147890753
// MI455X (gfx1250) — compile-verified, matrix operand order checked by reading, not run on hardware
//
#include <hip/hip_runtime.h>
#include <math.h>

// CDNA5 / gfx1250: wave32, WMMA 16x16x32 f16 -> f32
typedef __attribute__((ext_vector_type(16))) _Float16 v16h;
typedef __attribute__((ext_vector_type(8)))  float    v8f;
typedef __attribute__((ext_vector_type(4)))  float    v4f;

#define WAVES_PER_BLOCK 4

// Generic batched GEMM: C = max(A @ B + bias*biasScale, reluFloor)
// A: [M x K] row-major (lda), B: [K x N] row-major (ldb), C: [M x N] (ldc), f32 in/out.
// bias must be a valid pointer (use biasScale = 0 to disable).
// reluFloor: 0.0f for relu, -INFINITY for identity (branchless epilogue).
// Batch offsets: A += z*sAz + y*sAy, etc. One wave computes one 16x16 tile of C.
__global__ void wmma_gemm_f16(const float* __restrict__ A,
                              const float* __restrict__ B,
                              const float* __restrict__ bias,
                              float* __restrict__ C,
                              int M, int N, int K,
                              int lda, int ldb, int ldc,
                              long long sAz, long long sAy,
                              long long sBz, long long sBy,
                              long long sCz, long long sCy,
                              float biasScale, float reluFloor)
{
    const int lane = threadIdx.x & 31;
    const int wave = threadIdx.x >> 5;
    const int Ntiles = N >> 4;
    const int Mtiles = M >> 4;
    const int tile = blockIdx.x * WAVES_PER_BLOCK + wave;
    if (tile >= Mtiles * Ntiles) return;   // wave-uniform exit: EXEC stays all-ones
    const int tM = tile / Ntiles;
    const int tN = tile % Ntiles;

    A += (size_t)blockIdx.z * sAz + (size_t)blockIdx.y * sAy;
    B += (size_t)blockIdx.z * sBz + (size_t)blockIdx.y * sBy;
    C += (size_t)blockIdx.z * sCz + (size_t)blockIdx.y * sCy;

    const int m  = tM * 16 + (lane & 15);
    const int n  = tN * 16 + (lane & 15);
    const int hi = (lane >> 4) & 1;        // lane group: 0 => lanes 0-15, 1 => lanes 16-31
    const int kbase = hi << 3;             // ISA 16-bit A/B fragment K mapping:
                                           // lanes 0-15:  h 0..7 -> K 0..7,  h 8..15 -> K 16..23
                                           // lanes 16-31: h 0..7 -> K 8..15, h 8..15 -> K 24..31
    const float* Arow = A + (size_t)m * lda + kbase;   // 32B-aligned (lda, kbase multiples of 8/0)

    v8f acc = {};
    for (int kk = 0; kk < K; kk += 32) {
        // A fragment: two contiguous 8-float runs -> four b128 loads
        const v4f a0 = *(const v4f*)(Arow + kk);
        const v4f a1 = *(const v4f*)(Arow + kk + 4);
        const v4f a2 = *(const v4f*)(Arow + kk + 16);
        const v4f a3 = *(const v4f*)(Arow + kk + 20);
        v16h a, b;
        #pragma unroll
        for (int q = 0; q < 4; ++q) {
            a[q]      = (_Float16)a0[q];
            a[q + 4]  = (_Float16)a1[q];
            a[q + 8]  = (_Float16)a2[q];
            a[q + 12] = (_Float16)a3[q];
        }
        // B fragment: lane-coalesced strided loads (n contiguous across lanes)
        #pragma unroll
        for (int h = 0; h < 16; ++h) {
            const int ko = kbase + ((h < 8) ? h : (h + 8));
            b[h] = (_Float16)B[(size_t)(kk + ko) * ldb + n];
        }
        acc = __builtin_amdgcn_wmma_f32_16x16x32_f16(
                  /*neg_a=*/false, a, /*neg_b=*/false, b,
                  /*c_mod=*/(short)0, acc, /*reuse_a=*/false, /*reuse_b=*/false);
    }

    const float bv = bias[n] * biasScale;
    float* Ccol = C + (size_t)(tM * 16 + kbase) * ldc + n;
    #pragma unroll
    for (int v = 0; v < 8; ++v) {
        // C/D layout: VGPR v holds row tM*16 + v (lanes 0-15) or + v+8 (lanes 16-31)
        Ccol[(size_t)v * ldc] = fmaxf(acc[v] + bv, reluFloor);
    }
}

// Asum[t*10+w][j] = sum_k relu(P0[t][i0k][j] + P1[t][i1k][j] + P2[t][i2k][j] + b1[j])
// P: [16][3][160][1024] f32, idx: [16][10][3][512] int32, fully coalesced row reads (L2 resident).
__global__ void gather_relu_sum(const float* __restrict__ P,
                                const int*   __restrict__ idx,
                                const float* __restrict__ b1,
                                float*       __restrict__ Asum)
{
    const int tw = blockIdx.x;         // 0..159
    const int t  = tw / 10;
    const float* Pt = P + (size_t)t * 3u * 160u * 1024u;
    const int*   id = idx + (size_t)tw * 3u * 512u;

    const int j = threadIdx.x;         // 256 threads, 4 columns each
    float bb[4], acc[4];
    #pragma unroll
    for (int q = 0; q < 4; ++q) { bb[q] = b1[j + 256 * q]; acc[q] = 0.0f; }

    for (int k = 0; k < 512; ++k) {
        const int i0 = id[k];
        const int i1 = id[512 + k];
        const int i2 = id[1024 + k];
        const float* ra = Pt                        + (size_t)i0 * 1024u;
        const float* rb = Pt + (size_t)160u * 1024u + (size_t)i1 * 1024u;
        const float* rc = Pt + (size_t)320u * 1024u + (size_t)i2 * 1024u;
        #pragma unroll
        for (int q = 0; q < 4; ++q) {
            const int jj = j + 256 * q;
            const float v = ra[jj] + rb[jj] + rc[jj] + bb[q];
            acc[q] += fmaxf(v, 0.0f);
        }
    }
    #pragma unroll
    for (int q = 0; q < 4; ++q)
        Asum[(size_t)tw * 1024u + j + 256 * q] = acc[q];
}

// Row softmax over 64 columns; one block (64 threads) per row.
__global__ void softmax64(const float* __restrict__ score, float* __restrict__ out)
{
    __shared__ float red[64];
    const int r = blockIdx.x, j = threadIdx.x;
    const float v = score[(size_t)r * 64u + j];

    red[j] = v; __syncthreads();
    for (int s = 32; s > 0; s >>= 1) {
        if (j < s) red[j] = fmaxf(red[j], red[j + s]);
        __syncthreads();
    }
    const float mx = red[0]; __syncthreads();

    const float e = expf(v - mx);
    red[j] = e; __syncthreads();
    for (int s = 32; s > 0; s >>= 1) {
        if (j < s) red[j] += red[j + s];
        __syncthreads();
    }
    out[(size_t)r * 64u + j] = e / red[0];
}

extern "C" void kernel_launch(void* const* d_in, const int* in_sizes, int n_in,
                              void* d_out, int out_size, void* d_ws, size_t ws_size,
                              hipStream_t stream)
{
    const float* x   = (const float*)d_in[0];   // (16,16,10,512) == (16,160,512)
    const int*   tri = (const int*)  d_in[1];   // (16,10,3,512)
    const float* W1  = (const float*)d_in[2];   // (1536,1024)
    const float* b1  = (const float*)d_in[3];   // (1024)
    const float* W2  = (const float*)d_in[4];   // (1024,512)
    const float* b2  = (const float*)d_in[5];   // (512)
    const float* W3  = (const float*)d_in[6];   // (512,1024)
    const float* b3  = (const float*)d_in[7];   // (1024)
    const float* W4  = (const float*)d_in[8];   // (1024,512)
    const float* b4  = (const float*)d_in[9];   // (512)
    const float* Wc  = (const float*)d_in[10];  // (512,64)
    const float* bc  = (const float*)d_in[11];  // (64)
    float* out = (float*)d_out;                 // (16,10,64)

    const float NEG_INF = -INFINITY;

    // Workspace carve-up (all f32): ~33.5 MB
    float* P    = (float*)d_ws;                     // [16][3][160][1024]
    float* Asum = P    + (size_t)16 * 3 * 160 * 1024;
    float* g    = Asum + (size_t)160 * 1024;
    float* u    = g    + (size_t)160 * 512;
    float* o    = u    + (size_t)160 * 1024;
    float* sc   = o    + (size_t)160 * 512;

    // 1) P[t][p] = X_t @ W1[p*512:(p+1)*512]   (M=160, N=1024, K=512; batch z=t, y=p)
    //    bias disabled via scale 0 (pointer must stay valid).
    wmma_gemm_f16<<<dim3(160, 3, 16), 128, 0, stream>>>(
        x, W1, b1, P,
        160, 1024, 512, 512, 1024, 1024,
        (long long)160 * 512, 0LL,           // A: per-t
        0LL, (long long)512 * 1024,          // B: per-p row block of W1
        (long long)3 * 160 * 1024, (long long)160 * 1024,  // C: [t][p]
        0.0f, NEG_INF);

    // 2) Asum[t,w] = sum_k relu(P0[i0]+P1[i1]+P2[i2]+b1)
    gather_relu_sum<<<dim3(160), 256, 0, stream>>>(P, tri, b1, Asum);

    // 3) g = Asum @ W2 + 512*b2   (sum-through-linear: K=512 copies of b2)
    wmma_gemm_f16<<<dim3(80), 128, 0, stream>>>(
        Asum, W2, b2, g, 160, 512, 1024, 1024, 512, 512,
        0LL, 0LL, 0LL, 0LL, 0LL, 0LL, 512.0f, NEG_INF);

    // 4) u = relu(g @ W3 + b3)
    wmma_gemm_f16<<<dim3(160), 128, 0, stream>>>(
        g, W3, b3, u, 160, 1024, 512, 512, 1024, 1024,
        0LL, 0LL, 0LL, 0LL, 0LL, 0LL, 1.0f, 0.0f);

    // 5) o = u @ W4 + b4
    wmma_gemm_f16<<<dim3(80), 128, 0, stream>>>(
        u, W4, b4, o, 160, 512, 1024, 1024, 512, 512,
        0LL, 0LL, 0LL, 0LL, 0LL, 0LL, 1.0f, NEG_INF);

    // 6) score = o @ Wc + bc
    wmma_gemm_f16<<<dim3(10), 128, 0, stream>>>(
        o, Wc, bc, sc, 160, 64, 512, 512, 64, 64,
        0LL, 0LL, 0LL, 0LL, 0LL, 0LL, 1.0f, NEG_INF);

    // 7) softmax over the 64 classes
    softmax64<<<dim3(160), 64, 0, stream>>>(sc, out);
}
